// UniMolEncoder_31636729102633
// MI455X (gfx1250) — matrix operand order repaired and checked by reading, not compiled, not run on hardware
//
#include <hip/hip_runtime.h>
#include <hip/hip_bf16.h>
#include <cstdint>

#define ASYNC_STAGE 1   // use global_load_async_to_lds_b128 for aligned staging

// ---------------- problem constants ----------------
constexpr int BC = 16, NC = 160, N1C = 161, DC = 768, PC = 64, KC = 128;
constexpr int HC = 16, DhC = 48, FC = 3072, LC = 2;
constexpr int NFEATC = 9;
constexpr int MC  = BC * N1C;                         // 2576 tokens (incl CLS)
constexpr int NNC = NC * NC;                          // 25600 pairs per batch
constexpr long long MN1C = (long long)BC * N1C * N1C; // 414736 pair rows
constexpr int BNC = BC * NC;                          // 2560 atoms
constexpr int PLD = 176;                              // padded probs row stride (16-elem mult)

typedef __attribute__((ext_vector_type(16))) __bf16         v16bf;
typedef __attribute__((ext_vector_type(16))) unsigned short v16u;
typedef __attribute__((ext_vector_type(8)))  float          v8f;

enum { E_F32 = 0, E_BF16 = 1, E_GELU_BF16 = 2, E_ADD = 3, E_SCORES = 4, E_PAIR = 5 };

struct GemmEpiAux {
  const float* bias;       // per-column bias (nullable)
  const float* pb;         // [B,N1,N1,H] pair bias   (E_SCORES)
  const float* attn_bias;  // [B,N1,N1]               (E_SCORES)
  const float* atom_mask;  // [B,N]                   (E_SCORES)
  float scale;             // 1/sqrt(Dh)              (E_SCORES)
};

__device__ __forceinline__ float gelu_tanh(float x) {
  float u = 0.7978845608028654f * (x + 0.044715f * x * x * x);
  return 0.5f * x * (1.0f + tanhf(u));
}

__device__ __forceinline__ void async_ld_b128(unsigned ldsoff, const void* gptr) {
#if ASYNC_STAGE
  asm volatile("global_load_async_to_lds_b128 %0, %1, off" :: "v"(ldsoff), "v"(gptr) : "memory");
#else
  (void)ldsoff; (void)gptr;
#endif
}
__device__ __forceinline__ void async_wait() {
#if ASYNC_STAGE
  asm volatile("s_wait_asynccnt 0x0" ::: "memory");
#endif
}

// ---------------------------------------------------------------------------
// Batched bf16 GEMM, C(f32 acc) = A[MxK] * B[KxN]  (TRANSB: B stored [N x K]).
// WMMA 16x16x32 bf16. Block = 256 thr = 8 waves as 4(M) x 2(N); each wave owns
// a 16(M) x 32(N) slab -> 2 accumulators. Block tile 64x64.
// Double-buffered LDS; aligned tiles staged via global_load_async_to_lds_b128.
// Out-of-range M/N rows are CLAMPED (their outputs are masked in the epilogue);
// only K-tail elements must be zeroed (branchless select, no exec branching).
// LDS: At[m][k] row-major, Bt[n][k] n-major -> every fragment = 2x ds_load_b128.
// ---------------------------------------------------------------------------
template<int EPI, bool TRANSB>
__global__ __launch_bounds__(256)
void gemm_bf16_kernel(const __hip_bfloat16* __restrict__ Ag, long long lda, long long sAo, long long sAi,
                      const __hip_bfloat16* __restrict__ Bg, long long ldb, long long sBo, long long sBi,
                      void* __restrict__ Cg, long long ldc, long long sCo, long long sCi,
                      int M, int N, int Kd, int innerCount, GemmEpiAux aux) {
  __shared__ alignas(16) unsigned short At[2][64][48];   // 64 rows x 32 K (stride 48 = 96B)
  __shared__ alignas(16) unsigned short Bt[2][64][48];   // 64 cols(n) x 32 K
  const int tid  = threadIdx.x;
  const int wave = tid >> 5, lane = tid & 31;
  const int half = lane >> 4, lr = lane & 15;
  const int mi = wave >> 1, nj = wave & 1;
  const int z = blockIdx.z;
  const int outer = z / innerCount, inner = z % innerCount;
  const unsigned short* A  = reinterpret_cast<const unsigned short*>(Ag) + outer * sAo + inner * sAi;
  const unsigned short* Bm = reinterpret_cast<const unsigned short*>(Bg) + outer * sBo + inner * sBi;
  const int m0 = blockIdx.x * 64;
  const int n0 = blockIdx.y * 64;
  const bool alignA = ((lda & 7) == 0) && ((((uintptr_t)A) & 15) == 0);
  const bool alignB = ((ldb & 7) == 0) && ((((uintptr_t)Bm) & 15) == 0);
  v8f acc0 = {0.f, 0.f, 0.f, 0.f, 0.f, 0.f, 0.f, 0.f};
  v8f acc1 = {0.f, 0.f, 0.f, 0.f, 0.f, 0.f, 0.f, 0.f};
  union Frag { uint4 q[2]; v16u u; v16bf v; };

  auto stageA = [&](int buf, int k0) {
    if (alignA && (k0 + 32 <= Kd)) {
      int row = tid >> 2, seg = tid & 3;                       // 1 x b128 per thread
      int gm = m0 + row; gm = (gm < M) ? gm : (M - 1);         // clamp (output masked)
      const void* gp = A + (size_t)gm * lda + k0 + seg * 8;
#if ASYNC_STAGE
      async_ld_b128((unsigned)(uintptr_t)&At[buf][row][seg * 8], gp);
#else
      *reinterpret_cast<uint4*>(&At[buf][row][seg * 8]) = *reinterpret_cast<const uint4*>(gp);
#endif
    } else {                                                   // branchless clamped tail
      unsigned short tmp[8];
      #pragma unroll
      for (int q = 0; q < 8; ++q) {
        int l = tid + q * 256, r = l >> 5, c = l & 31;
        int gm = m0 + r;  gm = (gm < M) ? gm : (M - 1);
        int gk = k0 + c;  int gkc = (gk < Kd) ? gk : (Kd - 1);
        unsigned short t = A[(long long)gm * lda + gkc];
        tmp[q] = (gk < Kd) ? t : (unsigned short)0;
      }
      #pragma unroll
      for (int q = 0; q < 8; ++q) { int l = tid + q * 256; At[buf][l >> 5][l & 31] = tmp[q]; }
    }
  };

  auto stageB = [&](int buf, int k0) {
    if constexpr (TRANSB) {
      if (alignB && (k0 + 32 <= Kd)) {
        int n = tid >> 2, seg = tid & 3;
        int gn = n0 + n; gn = (gn < N) ? gn : (N - 1);          // clamp
        const void* gp = Bm + (size_t)gn * ldb + k0 + seg * 8;
#if ASYNC_STAGE
        async_ld_b128((unsigned)(uintptr_t)&Bt[buf][n][seg * 8], gp);
#else
        *reinterpret_cast<uint4*>(&Bt[buf][n][seg * 8]) = *reinterpret_cast<const uint4*>(gp);
#endif
      } else {
        unsigned short tmp[8];
        #pragma unroll
        for (int q = 0; q < 8; ++q) {
          int l = tid + q * 256, n = l >> 5, k = l & 31;
          int gn = n0 + n;  gn = (gn < N) ? gn : (N - 1);
          int gk = k0 + k;  int gkc = (gk < Kd) ? gk : (Kd - 1);
          unsigned short t = Bm[(long long)gn * ldb + gkc];
          tmp[q] = (gk < Kd) ? t : (unsigned short)0;
        }
        #pragma unroll
        for (int q = 0; q < 8; ++q) { int l = tid + q * 256; Bt[buf][l >> 5][l & 31] = tmp[q]; }
      }
    } else {
      if (alignB && (k0 + 32 <= Kd) && (n0 + 64 <= N)) {
        int k = tid >> 3, seg = tid & 7;                        // b128 load, scatter-transpose
        uint4 vv = *reinterpret_cast<const uint4*>(Bm + (size_t)(k0 + k) * ldb + n0 + seg * 8);
        const unsigned short* pv = reinterpret_cast<const unsigned short*>(&vv);
        #pragma unroll
        for (int q = 0; q < 8; ++q) Bt[buf][seg * 8 + q][k] = pv[q];
      } else {
        unsigned short tmp[8];
        #pragma unroll
        for (int q = 0; q < 8; ++q) {
          int l = tid + q * 256, n = l >> 5, k = l & 31;
          int gn = n0 + n;  int gnc = (gn < N) ? gn : (N - 1);
          int gk = k0 + k;  int gkc = (gk < Kd) ? gk : (Kd - 1);
          unsigned short t = Bm[(long long)gkc * ldb + gnc];
          tmp[q] = (gk < Kd) ? t : (unsigned short)0;
        }
        #pragma unroll
        for (int q = 0; q < 8; ++q) { int l = tid + q * 256; Bt[buf][l >> 5][l & 31] = tmp[q]; }
      }
    }
  };

  int cur = 0;
  stageA(0, 0);
  stageB(0, 0);
  for (int k0 = 0; k0 < Kd; k0 += 32) {
    async_wait();            // current buffer's async DMA complete (this wave)
    __syncthreads();         // all waves' stores visible; prior reads done
    int kn = k0 + 32;
    if (kn < Kd) { stageA(cur ^ 1, kn); stageB(cur ^ 1, kn); }  // prefetch next tile
    Frag fa, fb0, fb1;
    {
      const uint4* ap = reinterpret_cast<const uint4*>(&At[cur][mi * 16 + lr][0]);
      fa.q[0] = ap[half * 2];      // K = half*16 .. +7
      fa.q[1] = ap[half * 2 + 1];  // K = half*16+8 .. +15
      const uint4* bp0 = reinterpret_cast<const uint4*>(&Bt[cur][nj * 32 + lr][0]);
      fb0.q[0] = bp0[half * 2];    // K = half*16 .. +7
      fb0.q[1] = bp0[half * 2 + 1];
      const uint4* bp1 = reinterpret_cast<const uint4*>(&Bt[cur][nj * 32 + 16 + lr][0]);
      fb1.q[0] = bp1[half * 2];
      fb1.q[1] = bp1[half * 2 + 1];
    }
    acc0 = __builtin_amdgcn_wmma_f32_16x16x32_bf16(false, fa.v, false, fb0.v, (short)0, acc0, false, false);
    acc1 = __builtin_amdgcn_wmma_f32_16x16x32_bf16(false, fa.v, false, fb1.v, (short)0, acc1, false, false);
    cur ^= 1;
  }

  // ---------------- epilogue ----------------
  #pragma unroll
  for (int s = 0; s < 2; ++s) {
    v8f acc = s ? acc1 : acc0;
    #pragma unroll
    for (int i = 0; i < 8; ++i) {
      int gm = m0 + mi * 16 + half * 8 + i;    // C: vgpr i -> row i+half*8 in 16x16 tile
      int gn = n0 + nj * 32 + s * 16 + lr;
      if (gm >= M || gn >= N) continue;
      float v = acc[i];
      if (aux.bias) v += aux.bias[gn];
      long long coff = outer * sCo + inner * sCi + (long long)gm * ldc + gn;
      if constexpr (EPI == E_F32) {
        ((float*)Cg)[coff] = v;
      } else if constexpr (EPI == E_BF16) {
        ((__hip_bfloat16*)Cg)[coff] = __float2bfloat16(v);
      } else if constexpr (EPI == E_GELU_BF16) {
        ((__hip_bfloat16*)Cg)[coff] = __float2bfloat16(gelu_tanh(v));
      } else if constexpr (EPI == E_ADD) {
        ((float*)Cg)[coff] += v;                               // residual add
      } else if constexpr (EPI == E_SCORES) {
        int b = outer, h = inner;
        long long ij = ((long long)b * N1C + gm) * N1C + gn;
        float mci = (gm == 0) ? 1.f : aux.atom_mask[b * NC + gm - 1];
        float mcj = (gn == 0) ? 1.f : aux.atom_mask[b * NC + gn - 1];
        float neg = (mci * mcj > 0.f) ? 0.f : -1e9f;
        ((float*)Cg)[coff] = v * aux.scale + aux.pb[ij * HC + h] + aux.attn_bias[ij] + neg;
      } else if constexpr (EPI == E_PAIR) {
        int ii = gm / NC, jj = gm % NC;                        // interior accumulate
        ((float*)Cg)[outer * sCo + ((long long)(ii + 1) * N1C + (jj + 1)) * PC + gn] += v;
      }
    }
  }
}

// ---------------- elementwise / reduction kernels ----------------
__global__ void f2bf_kernel(const float* __restrict__ in, __hip_bfloat16* __restrict__ out, long long n) {
  long long i = (long long)blockIdx.x * blockDim.x + threadIdx.x;
  if (i < n) out[i] = __float2bfloat16(in[i]);
}

__global__ void f2bf_transpose_kernel(const float* __restrict__ in, __hip_bfloat16* __restrict__ out,
                                      int rows, int cols) {  // in[rows][cols] -> out[cols][rows]
  int i = blockIdx.x * blockDim.x + threadIdx.x;
  if (i < rows * cols) {
    int r = i / cols, c = i % cols;
    out[(long long)c * rows + r] = __float2bfloat16(in[i]);
  }
}

__global__ void embed_kernel(const int* __restrict__ tok, const int* __restrict__ afeat,
                             const int* __restrict__ deg, const float* __restrict__ tok_emb,
                             const float* __restrict__ af_emb, const float* __restrict__ deg_emb,
                             const float* __restrict__ cls, float* __restrict__ x) {
  int bn = blockIdx.x;                 // 0..B*N1-1
  int b = bn / N1C, n = bn % N1C;
  float* xr = x + (long long)bn * DC;
  if (n == 0) { for (int c = threadIdx.x; c < DC; c += blockDim.x) xr[c] = cls[c]; return; }
  int an = n - 1;
  int t  = tok[b * NC + an];
  int dg = deg[b * NC + an];
  int fv[NFEATC];
  #pragma unroll
  for (int f = 0; f < NFEATC; ++f) fv[f] = afeat[(b * NC + an) * NFEATC + f];
  for (int c = threadIdx.x; c < DC; c += blockDim.x) {
    float v = tok_emb[(long long)t * DC + c] + deg_emb[(long long)dg * DC + c];
    #pragma unroll
    for (int f = 0; f < NFEATC; ++f) v += af_emb[((long long)f * 16 + fv[f]) * DC + c];
    xr[c] = v;
  }
}

__global__ void pair_init_kernel(const int* __restrict__ sp, const int* __restrict__ ef,
                                 const float* __restrict__ sp_emb, const float* __restrict__ ef_emb,
                                 float* __restrict__ pair) {
  long long idx = blockIdx.x;          // B*N1*N1
  int b = (int)(idx / ((long long)N1C * N1C));
  int rem = (int)(idx % ((long long)N1C * N1C));
  int i = rem / N1C, j = rem % N1C;
  int p = threadIdx.x;                 // 64
  float* pr = pair + idx * PC;
  if (i == 0 || j == 0) { pr[p] = 0.f; return; }
  long long e = ((long long)b * NC + (i - 1)) * NC + (j - 1);
  float v = sp_emb[(long long)sp[e] * PC + p];
  #pragma unroll
  for (int t = 0; t < 3; ++t) v += ef_emb[((long long)t * 64 + ef[e * 3 + t]) * PC + p];
  pr[p] = v;
}

__global__ void gauss_kernel(int b, const float* __restrict__ coord, const int* __restrict__ ptype,
                             const float* __restrict__ mul, const float* __restrict__ bias,
                             const float* __restrict__ means, const float* __restrict__ stds,
                             __hip_bfloat16* __restrict__ g) {
  int ij = blockIdx.x;                 // NN
  int i = ij / NC, j = ij % NC;
  const float* ci = coord + ((long long)b * NC + i) * 3;
  const float* cj = coord + ((long long)b * NC + j) * 3;
  float dx = ci[0] - cj[0], dy = ci[1] - cj[1], dz = ci[2] - cj[2];
  float dist = sqrtf(dx * dx + dy * dy + dz * dz);
  int pt = ptype[((long long)b * NC + i) * NC + j];
  float xg = dist * mul[pt] + bias[pt];
  int k = threadIdx.x;                 // 128
  float zz = (xg - means[k]) / stds[k];
  float val = __expf(-0.5f * zz * zz) / (stds[k] * 2.5066282746310002f);
  g[(long long)ij * KC + k] = __float2bfloat16(val);
}

template<bool BF16OUT>
__global__ __launch_bounds__(256)
void ln_rows_kernel(const float* __restrict__ x, const float* __restrict__ g,
                    const float* __restrict__ bta, int cols,
                    float* __restrict__ outF, __hip_bfloat16* __restrict__ outB) {
  __shared__ float red[256];
  long long row = blockIdx.x;
  const float* xr = x + row * cols;
  float v[8]; int cnt = 0; float s = 0.f;
  for (int c = threadIdx.x; c < cols; c += 256) { float t = xr[c]; v[cnt++] = t; s += t; }
  red[threadIdx.x] = s; __syncthreads();
  for (int o = 128; o > 0; o >>= 1) { if (threadIdx.x < o) red[threadIdx.x] += red[threadIdx.x + o]; __syncthreads(); }
  float mean = red[0] / cols; __syncthreads();
  float s2 = 0.f;
  for (int q = 0; q < cnt; ++q) { float d = v[q] - mean; s2 += d * d; }
  red[threadIdx.x] = s2; __syncthreads();
  for (int o = 128; o > 0; o >>= 1) { if (threadIdx.x < o) red[threadIdx.x] += red[threadIdx.x + o]; __syncthreads(); }
  float rstd = rsqrtf(red[0] / cols + 1e-5f);
  cnt = 0;
  for (int c = threadIdx.x; c < cols; c += 256) {
    float o = (v[cnt++] - mean) * rstd * g[c] + bta[c];
    if (BF16OUT) outB[row * cols + c] = __float2bfloat16(o);
    else         outF[row * cols + c] = o;
  }
}

__global__ __launch_bounds__(256)
void ln_pair64_kernel(const float* __restrict__ pair, const float* __restrict__ g,
                      const float* __restrict__ bta, long long rows,
                      __hip_bfloat16* __restrict__ out) {
  long long row = (long long)blockIdx.x * 8 + (threadIdx.x >> 5);
  if (row >= rows) return;
  int lane = threadIdx.x & 31;
  const float* pr = pair + row * 64;
  float a0 = pr[lane], a1 = pr[lane + 32];
  float s = a0 + a1;
  for (int o = 16; o > 0; o >>= 1) s += __shfl_xor(s, o, 32);
  float mean = s * (1.f / 64.f);
  float d0 = a0 - mean, d1 = a1 - mean;
  float s2 = d0 * d0 + d1 * d1;
  for (int o = 16; o > 0; o >>= 1) s2 += __shfl_xor(s2, o, 32);
  float rstd = rsqrtf(s2 * (1.f / 64.f) + 1e-5f);
  out[row * 64 + lane]      = __float2bfloat16(d0 * rstd * g[lane] + bta[lane]);
  out[row * 64 + lane + 32] = __float2bfloat16(d1 * rstd * g[lane + 32] + bta[lane + 32]);
}

__global__ __launch_bounds__(256)
void softmax_rows_kernel(const float* __restrict__ scores, long long rows,
                         __hip_bfloat16* __restrict__ probs) {
  long long row = (long long)blockIdx.x * 8 + (threadIdx.x >> 5);
  if (row >= rows) return;
  int lane = threadIdx.x & 31;
  const float* sr = scores + row * N1C;
  float v[6]; int cnt = 0; float mx = -3.4e38f;
  for (int c = lane; c < N1C; c += 32) { float t = sr[c]; v[cnt++] = t; mx = fmaxf(mx, t); }
  for (int o = 16; o > 0; o >>= 1) mx = fmaxf(mx, __shfl_xor(mx, o, 32));
  float s = 0.f;
  for (int q = 0; q < cnt; ++q) { v[q] = __expf(v[q] - mx); s += v[q]; }
  for (int o = 16; o > 0; o >>= 1) s += __shfl_xor(s, o, 32);
  float inv = 1.f / s;
  cnt = 0;
  for (int c = lane; c < N1C; c += 32) probs[row * PLD + c] = __float2bfloat16(v[cnt++] * inv);
}

__global__ void pair_update_kernel(float* __restrict__ pair, const __hip_bfloat16* __restrict__ a,
                                   const __hip_bfloat16* __restrict__ bb,
                                   const float* __restrict__ atom_mask) {
  long long idx = blockIdx.x;
  int b = (int)(idx / ((long long)N1C * N1C));
  int rem = (int)(idx % ((long long)N1C * N1C));
  int i = rem / N1C, j = rem % N1C;
  float mi = (i == 0) ? 1.f : atom_mask[b * NC + i - 1];
  float mj = (j == 0) ? 1.f : atom_mask[b * NC + j - 1];
  float pm = mi * mj;
  int p = threadIdx.x;
  float av = __bfloat162float(a[((long long)b * N1C + i) * PC + p]);
  float bv = __bfloat162float(bb[((long long)b * N1C + j) * PC + p]);
  pair[idx * PC + p] += av * bv * pm;
}

__global__ __launch_bounds__(256)
void mask_scan_kernel(const float* __restrict__ mask, int* __restrict__ idx) {
  __shared__ int part[256]; __shared__ int tot;
  int t = threadIdx.x;
  const int per = (BNC + 255) / 256;   // 10
  int base = t * per, c = 0;
  for (int q = 0; q < per; ++q) { int r = base + q; if (r < BNC && mask[r] > 0.f) c++; }
  part[t] = c; __syncthreads();
  if (t == 0) { int run = 0; for (int q = 0; q < 256; ++q) { int tmp = part[q]; part[q] = run; run += tmp; } tot = run; }
  __syncthreads();
  int pos = part[t];
  for (int q = 0; q < per; ++q) { int r = base + q; if (r < BNC && mask[r] > 0.f) idx[pos++] = r; }
  __syncthreads();
  for (int r = tot + t; r < BNC; r += 256) idx[r] = 0;   // jnp.nonzero size-pad fill
}

__global__ void gather_out_kernel(const float* __restrict__ x, const int* __restrict__ idx,
                                  float* __restrict__ out) {
  long long i = (long long)blockIdx.x * blockDim.x + threadIdx.x;
  const long long atomsN = (long long)BNC * DC;
  const long long clsN   = (long long)BC * DC;
  if (i < atomsN) {
    int r = (int)(i / DC), c = (int)(i % DC);
    int src = idx[r];
    int sb = src / NC, sn = src % NC;
    out[i] = x[((long long)sb * N1C + 1 + sn) * DC + c];
  } else if (i < atomsN + clsN) {
    long long k = i - atomsN; int b = (int)(k / DC), c = (int)(k % DC);
    out[i] = x[((long long)b * N1C) * DC + c];
  } else if (i < atomsN + clsN + BNC) {
    int r = (int)(i - atomsN - clsN);
    out[i] = (float)(idx[r] / NC);
  }
}

// ---------------- host-side orchestration ----------------
#define LAUNCH_GEMM(EPI, TB, Aptr, lda, sAo, sAi, Bptr, ldb, sBo, sBi, Cptr, ldc, sCo, sCi, Mv, Nv, Kv, Zv, ICv, AUXv) \
  hipLaunchKernelGGL((gemm_bf16_kernel<EPI, TB>),                                                   \
    dim3((unsigned)(((Mv) + 63) / 64), (unsigned)(((Nv) + 63) / 64), (unsigned)(Zv)), dim3(256), 0, stream, \
    (const __hip_bfloat16*)(Aptr), (long long)(lda), (long long)(sAo), (long long)(sAi),            \
    (const __hip_bfloat16*)(Bptr), (long long)(ldb), (long long)(sBo), (long long)(sBi),            \
    (void*)(Cptr), (long long)(ldc), (long long)(sCo), (long long)(sCi),                            \
    (int)(Mv), (int)(Nv), (int)(Kv), (int)(ICv), (AUXv))

extern "C" void kernel_launch(void* const* d_in, const int* in_sizes, int n_in,
                              void* d_out, int out_size, void* d_ws, size_t ws_size,
                              hipStream_t stream) {
  (void)in_sizes; (void)n_in; (void)out_size; (void)ws_size;
  // inputs
  const int*   src_tokens = (const int*)  d_in[0];
  const float* atom_mask  = (const float*)d_in[1];
  const float* src_coord  = (const float*)d_in[2];
  const int*   atom_feat  = (const int*)  d_in[3];
  const int*   degree     = (const int*)  d_in[4];
  const float* attn_bias  = (const float*)d_in[5];
  const int*   shortest   = (const int*)  d_in[6];
  const int*   edge_feat  = (const int*)  d_in[7];
  const int*   pair_type  = (const int*)  d_in[8];
  const float* tok_emb = (const float*)d_in[9];
  const float* af_emb  = (const float*)d_in[10];
  const float* deg_emb = (const float*)d_in[11];
  const float* cls     = (const float*)d_in[12];
  const float* sp_emb  = (const float*)d_in[13];
  const float* ef_emb  = (const float*)d_in[14];
  const float* gbf_mul   = (const float*)d_in[15];
  const float* gbf_bias  = (const float*)d_in[16];
  const float* gbf_means = (const float*)d_in[17];
  const float* gbf_stds  = (const float*)d_in[18];
  const float* gbf_w1 = (const float*)d_in[19];
  const float* gbf_b1 = (const float*)d_in[20];
  const float* gbf_w2 = (const float*)d_in[21];
  const float* gbf_b2 = (const float*)d_in[22];
  auto LPRM = [&](int layer, int off) { return (const float*)d_in[23 + layer * 19 + off]; };
  const float* lnf_g = (const float*)d_in[61];
  const float* lnf_b = (const float*)d_in[62];

  // workspace arena
  size_t off = 0;
  auto alloc = [&](size_t bytes) -> char* {
    char* p = (char*)d_ws + off;
    off += (bytes + 255) & ~(size_t)255;
    return p;
  };
  float* x      = (float*)alloc(sizeof(float) * (size_t)MC * DC);
  float* pair   = (float*)alloc(sizeof(float) * (size_t)MN1C * PC);
  float* pb     = (float*)alloc(sizeof(float) * (size_t)MN1C * HC);
  float* scores = (float*)alloc(sizeof(float) * (size_t)BC * HC * N1C * N1C);
  __hip_bfloat16* pairln = (__hip_bfloat16*)alloc(2 * (size_t)MN1C * PC);
  __hip_bfloat16* probs  = (__hip_bfloat16*)alloc(2 * (size_t)BC * HC * N1C * PLD);
  __hip_bfloat16* hbf    = (__hip_bfloat16*)alloc(2 * (size_t)MC * DC);
  __hip_bfloat16* qbf    = (__hip_bfloat16*)alloc(2 * (size_t)MC * DC);
  __hip_bfloat16* kbf    = (__hip_bfloat16*)alloc(2 * (size_t)MC * DC);
  __hip_bfloat16* vbf    = (__hip_bfloat16*)alloc(2 * (size_t)MC * DC);
  __hip_bfloat16* obf    = (__hip_bfloat16*)alloc(2 * (size_t)MC * DC);
  __hip_bfloat16* t1bf   = (__hip_bfloat16*)alloc(2 * (size_t)MC * FC);
  __hip_bfloat16* wbf    = (__hip_bfloat16*)alloc(2 * (size_t)DC * FC);
  __hip_bfloat16* w1g    = (__hip_bfloat16*)alloc(2 * (size_t)KC * PC);
  __hip_bfloat16* w2g    = (__hip_bfloat16*)alloc(2 * (size_t)PC * PC);
  __hip_bfloat16* wpbT   = (__hip_bfloat16*)alloc(2 * (size_t)PC * HC);
  __hip_bfloat16* abf    = (__hip_bfloat16*)alloc(2 * (size_t)MC * PC);
  __hip_bfloat16* bbf    = (__hip_bfloat16*)alloc(2 * (size_t)MC * PC);
  __hip_bfloat16* gch    = (__hip_bfloat16*)alloc(2 * (size_t)NNC * KC);
  __hip_bfloat16* t1ch   = (__hip_bfloat16*)alloc(2 * (size_t)NNC * PC);
  int* idxbuf = (int*)alloc(sizeof(int) * BNC);

  auto conv = [&](const float* src, __hip_bfloat16* dst, long long n) {
    f2bf_kernel<<<(unsigned)((n + 255) / 256), 256, 0, stream>>>(src, dst, n);
  };
  GemmEpiAux axN{}; axN.bias = nullptr;

  // ---- phase 0: embeddings + pair init ----
  embed_kernel<<<MC, 256, 0, stream>>>(src_tokens, atom_feat, degree, tok_emb, af_emb, deg_emb, cls, x);
  pair_init_kernel<<<(unsigned)MN1C, 64, 0, stream>>>(shortest, edge_feat, sp_emb, ef_emb, pair);

  // ---- GBF gaussian MLP, chunked per batch b ----
  conv(gbf_w1, w1g, (long long)KC * PC);
  conv(gbf_w2, w2g, (long long)PC * PC);
  for (int b = 0; b < BC; ++b) {
    gauss_kernel<<<NNC, KC, 0, stream>>>(b, src_coord, pair_type, gbf_mul, gbf_bias,
                                         gbf_means, gbf_stds, gch);
    GemmEpiAux ax1{}; ax1.bias = gbf_b1;
    LAUNCH_GEMM(E_GELU_BF16, false, gch, KC, 0, 0, w1g, PC, 0, 0, t1ch, PC, 0, 0,
                NNC, PC, KC, 1, 1, ax1);
    GemmEpiAux ax2{}; ax2.bias = gbf_b2;
    LAUNCH_GEMM(E_PAIR, false, t1ch, PC, 0, 0, w2g, PC, 0, 0,
                pair + (size_t)b * N1C * N1C * PC, 0, 0, 0, NNC, PC, PC, 1, 1, ax2);
  }

  // ---- transformer layers ----
  const float inv_sqrt = 1.0f / sqrtf((float)DhC);
  for (int l = 0; l < LC; ++l) {
    // LN1 -> hbf
    ln_rows_kernel<true><<<MC, 256, 0, stream>>>(x, LPRM(l, 0), LPRM(l, 1), DC, nullptr, hbf);
    // Q, K, V projections (bf16 out)
    conv(LPRM(l, 2), wbf, (long long)DC * DC);
    LAUNCH_GEMM(E_BF16, false, hbf, DC, 0, 0, wbf, DC, 0, 0, qbf, DC, 0, 0, MC, DC, DC, 1, 1, axN);
    conv(LPRM(l, 3), wbf, (long long)DC * DC);
    LAUNCH_GEMM(E_BF16, false, hbf, DC, 0, 0, wbf, DC, 0, 0, kbf, DC, 0, 0, MC, DC, DC, 1, 1, axN);
    conv(LPRM(l, 4), wbf, (long long)DC * DC);
    LAUNCH_GEMM(E_BF16, false, hbf, DC, 0, 0, wbf, DC, 0, 0, vbf, DC, 0, 0, MC, DC, DC, 1, 1, axN);
    // pair LN (P=64) -> pairln ; pb = pairln @ Wpb  [MN1 x 16] via transposed weight
    ln_pair64_kernel<<<(unsigned)((MN1C + 7) / 8), 256, 0, stream>>>(pair, LPRM(l, 6), LPRM(l, 7), MN1C, pairln);
    f2bf_transpose_kernel<<<(PC * HC + 255) / 256, 256, 0, stream>>>(LPRM(l, 8), wpbT, PC, HC);
    LAUNCH_GEMM(E_F32, true, pairln, PC, 0, 0, wpbT, PC, 0, 0, pb, HC, 0, 0,
                (int)MN1C, HC, PC, 1, 1, axN);
    // scores = q k^T * inv_sqrt + pb + attn_bias + mask ; batched over (b,h)
    GemmEpiAux axS{}; axS.pb = pb; axS.attn_bias = attn_bias; axS.atom_mask = atom_mask; axS.scale = inv_sqrt;
    LAUNCH_GEMM(E_SCORES, true,
                qbf, DC, (long long)N1C * DC, DhC,
                kbf, DC, (long long)N1C * DC, DhC,
                scores, N1C, (long long)HC * N1C * N1C, (long long)N1C * N1C,
                N1C, N1C, DhC, BC * HC, HC, axS);
    softmax_rows_kernel<<<(unsigned)(((long long)BC * HC * N1C + 7) / 8), 256, 0, stream>>>(
        scores, (long long)BC * HC * N1C, probs);
    // o = probs @ v (bf16 out, scattered into [M, D] head slices)
    LAUNCH_GEMM(E_BF16, false,
                probs, PLD, (long long)HC * N1C * PLD, (long long)N1C * PLD,
                vbf, DC, (long long)N1C * DC, DhC,
                obf, DC, (long long)N1C * DC, DhC,
                N1C, DhC, N1C, BC * HC, HC, axN);
    // x += o @ Wo
    conv(LPRM(l, 5), wbf, (long long)DC * DC);
    LAUNCH_GEMM(E_ADD, false, obf, DC, 0, 0, wbf, DC, 0, 0, x, DC, 0, 0, MC, DC, DC, 1, 1, axN);
    // FFN
    ln_rows_kernel<true><<<MC, 256, 0, stream>>>(x, LPRM(l, 9), LPRM(l, 10), DC, nullptr, hbf);
    conv(LPRM(l, 11), wbf, (long long)DC * FC);
    GemmEpiAux axG{}; axG.bias = LPRM(l, 12);
    LAUNCH_GEMM(E_GELU_BF16, false, hbf, DC, 0, 0, wbf, FC, 0, 0, t1bf, FC, 0, 0, MC, FC, DC, 1, 1, axG);
    conv(LPRM(l, 13), wbf, (long long)FC * DC);
    GemmEpiAux axR{}; axR.bias = LPRM(l, 14);
    LAUNCH_GEMM(E_ADD, false, t1bf, FC, 0, 0, wbf, DC, 0, 0, x, DC, 0, 0, MC, DC, FC, 1, 1, axR);
    // pair update: a = ln3(x) @ Wa ; b = ln3(x) @ Wb ; pair += a_i * b_j * mask
    ln_rows_kernel<true><<<MC, 256, 0, stream>>>(x, LPRM(l, 15), LPRM(l, 16), DC, nullptr, hbf);
    conv(LPRM(l, 17), wbf, (long long)DC * PC);
    LAUNCH_GEMM(E_BF16, false, hbf, DC, 0, 0, wbf, PC, 0, 0, abf, PC, 0, 0, MC, PC, DC, 1, 1, axN);
    conv(LPRM(l, 18), wbf, (long long)DC * PC);
    LAUNCH_GEMM(E_BF16, false, hbf, DC, 0, 0, wbf, PC, 0, 0, bbf, PC, 0, 0, MC, PC, DC, 1, 1, axN);
    pair_update_kernel<<<(unsigned)MN1C, 64, 0, stream>>>(pair, abf, bbf, atom_mask);
  }

  // ---- final LN (in place) + masked gather ----
  ln_rows_kernel<false><<<MC, 256, 0, stream>>>(x, lnf_g, lnf_b, DC, x, nullptr);
  mask_scan_kernel<<<1, 256, 0, stream>>>(atom_mask, idxbuf);
  const long long outTot = (long long)BNC * DC + (long long)BC * DC + BNC;
  gather_out_kernel<<<(unsigned)((outTot + 255) / 256), 256, 0, stream>>>(x, idxbuf, (float*)d_out);
}
